// SpatiotemporalAttnEpiGNN_79602923864548
// MI455X (gfx1250) — compile-verified, matrix operand order checked by reading, not run on hardware
//
#include <hip/hip_runtime.h>
#include <hip/hip_bf16.h>
#include <math.h>

// ---------------- problem constants ----------------
#define B_    32
#define T_    32
#define M_    1024
#define F_    4
#define K_    8
#define HIDA_ 32
#define HIDR_ 40
#define NL_   3
#define TOUT_ 14

typedef unsigned short u16;
typedef unsigned int   u32;
typedef unsigned long long u64;
typedef __attribute__((ext_vector_type(16))) __bf16 v16bf;
typedef __attribute__((ext_vector_type(8)))  float  v8f;
typedef __attribute__((ext_vector_type(4)))  unsigned int v4u;
typedef __attribute__((ext_vector_type(4)))  int v4i;
typedef __attribute__((ext_vector_type(8)))  int v8i;

// ---------------- WMMA helpers ----------------
__device__ __forceinline__ v8f wmma_bf16(v16bf a, v16bf b, v8f c) {
  return __builtin_amdgcn_wmma_f32_16x16x32_bf16(
      false, a, false, b, (short)0, c, false, false);
}

// Load a 16x32 bf16 tile from row-major memory (global OR LDS) into the
// 16-bit A-operand layout (ISA 7.12.2): lane L holds row (L&15); lanes<16
// carry K={0..7,16..23}, lanes>=16 carry K={8..15,24..31}.  For D = X * Y^T
// the B operand is Y's rows loaded with this same pattern.
__device__ __forceinline__ v16bf load_a_tile(const u16* __restrict__ base, int ld) {
  const int lane = threadIdx.x & 31;
  const int row  = lane & 15;
  const int kb   = (lane & 16) ? 8 : 0;
  const u16* p = base + (size_t)row * ld + kb;
  union { uint4 q[2]; v16bf v; } u;
  u.q[0] = *(const uint4*)(p);        // K = kb .. kb+7
  u.q[1] = *(const uint4*)(p + 16);   // K = kb+16 .. kb+23
  return u.v;
}

__device__ __forceinline__ u16 f2bf(float f) {
  union { __bf16 h; u16 u; } cv; cv.h = (__bf16)f; return cv.u;
}

// ---------------- TDM: 1-D global -> LDS copy ----------------
// Builds a minimal D# (count=1, type=2, data_size=8B, 1-D tile) and issues
// TENSOR_LOAD_TO_LDS.  n8 = number of 8-byte elements (must be <= 65535).
// Caller: issue from one wave only, then s_wait_tensorcnt + block barrier.
__device__ __forceinline__ void tdm_load_1d(const void* gsrc, u32 lds_addr, u32 n8) {
#if __has_builtin(__builtin_amdgcn_tensor_load_to_lds)
  const u64 ga = (u64)gsrc;
  v4u g0;
  g0[0] = 1u;                                        // count=1, user descriptor
  g0[1] = lds_addr;                                  // LDS byte address
  g0[2] = (u32)ga;                                   // global_addr[31:0]
  g0[3] = (u32)((ga >> 32) & 0x1FFFFFFull) | (2u << 30);  // addr[56:32] | type=2
  v8i g1;
  g1[0] = (int)(3u << 16);                           // data_size = 8 bytes
  g1[1] = (int)((n8 & 0xFFFFu) << 16);               // tensor_dim0[15:0]
  g1[2] = (int)((n8 >> 16) | (1u << 16));            // tensor_dim0[31:16], tensor_dim1=1
  g1[3] = (int)((n8 & 0xFFFFu) << 16);               // tile_dim0 = n8
  g1[4] = 0;                                         // tile_dim1/2 unused
  g1[5] = (int)n8;                                   // tensor_dim0_stride[31:0]
  g1[6] = 0;
  g1[7] = 0;
  v4i z4 = {0, 0, 0, 0};
#if defined(__clang_major__) && __clang_major__ >= 23
  v8i z8 = {0, 0, 0, 0, 0, 0, 0, 0};                 // 6-arg toolchain variant
  __builtin_amdgcn_tensor_load_to_lds(g0, g1, z4, z4, z8, 0);
#else
  __builtin_amdgcn_tensor_load_to_lds(g0, g1, z4, z4, 0);   // ROCm 7.2 / clang-22
#endif
  __builtin_amdgcn_s_wait_tensorcnt(0);
#endif
}

// Fallback / generic cooperative copy (used when TDM builtin is unavailable).
__device__ __forceinline__ void coop_copy_8(void* dst, const void* src, int n8) {
  for (int t = threadIdx.x; t < n8; t += (int)blockDim.x)
    ((u64*)dst)[t] = ((const u64*)src)[t];
}

#if __has_builtin(__builtin_amdgcn_tensor_load_to_lds)
#define STAGE_LDS(dst, src, n8)                                       \
  do {                                                                \
    if ((threadIdx.x >> 5) == 0)                                      \
      tdm_load_1d((src), (u32)(size_t)(dst), (u32)(n8));              \
  } while (0)
#else
#define STAGE_LDS(dst, src, n8) coop_copy_8((dst), (src), (n8))
#endif

// ---------------- K0: conv branches -> emb (B,M,40) ----------------
struct ConvP { const float *w, *b, *g, *be, *mu, *var; };

__device__ __forceinline__ void conv_branch(const float4* __restrict__ xl,
                                            const ConvP p, int ks, int dil,
                                            float* __restrict__ out8) {
  const int outT = T_ - dil * (ks - 1);
  for (int k = 0; k < K_; ++k) {
    const float scale = p.g[k] * rsqrtf(p.var[k] + 1e-5f);
    const float mu = p.mu[k], shift = p.be[k], bias = p.b[k];
    float best = -3.0e38f;
    for (int t0 = 0; t0 < outT; ++t0) {
      float s = bias;
      for (int tau = 0; tau < ks; ++tau) {
        const float4 xv = xl[t0 + tau * dil];
        const float* wk = p.w + (size_t)k * F_ * ks + tau;   // w[k][c][tau]
        s += xv.x * wk[0] + xv.y * wk[ks] + xv.z * wk[2 * ks] + xv.w * wk[3 * ks];
      }
      const float y = (s - mu) * scale + shift;               // BN before pool
      best = fmaxf(best, y);
    }
    out8[k] = tanhf(best);                                    // branch tanh
  }
}

__global__ __launch_bounds__(128)
void k_emb(const float* __restrict__ x, ConvP c0, ConvP c1, ConvP c2, ConvP c3,
           ConvP c4, float* __restrict__ emb) {
  const int bm = blockIdx.x * blockDim.x + threadIdx.x;
  if (bm >= B_ * M_) return;
  const int b = bm / M_, m = bm % M_;
  float4 xl[T_];
  const float4* xg = (const float4*)x + (size_t)b * T_ * M_ + m;  // (b,t,m,:)
  for (int t = 0; t < T_; ++t) xl[t] = xg[(size_t)t * M_];
  float o[HIDR_];
  conv_branch(xl, c0, 3, 1, o + 0);
  conv_branch(xl, c1, 5, 1, o + 8);
  conv_branch(xl, c2, 3, 2, o + 16);
  conv_branch(xl, c3, 5, 2, o + 24);
  conv_branch(xl, c4, T_, 1, o + 32);
  float* er = emb + (size_t)bm * HIDR_;
  for (int j = 0; j < HIDR_; ++j) er[j] = tanhf(o[j]);        // outer tanh
}

// ---------------- K1: degrees + adjacency bitmask ----------------
__global__ __launch_bounds__(256)
void k_degree(const float* __restrict__ adj, float* __restrict__ dvec,
              u16* __restrict__ adjbits) {
  const int m = blockIdx.x * blockDim.x + threadIdx.x;
  if (m >= M_) return;
  float s = 0.f;
  for (int jw = 0; jw < M_ / 16; ++jw) {
    u32 bits = 0;
#pragma unroll
    for (int e = 0; e < 16; ++e) {
      const float a = adj[(size_t)m * M_ + jw * 16 + e];
      s += a;
      if (a != 0.f) bits |= (1u << e);
    }
    adjbits[m * (M_ / 16) + jw] = (u16)bits;
  }
  dvec[m] = s;
}

// ---------------- K2: q,k = emb@Wq/Wk + b  (bf16) ----------------
__global__ __launch_bounds__(256)
void k_qk(const float* __restrict__ emb,
          const float* __restrict__ wq_w, const float* __restrict__ wq_b,
          const float* __restrict__ wk_w, const float* __restrict__ wk_b,
          u16* __restrict__ qbf, u16* __restrict__ kbf) {
  const int bm = blockIdx.x * blockDim.x + threadIdx.x;
  if (bm >= B_ * M_) return;
  float e[HIDR_];
  const float* er = emb + (size_t)bm * HIDR_;
  for (int f = 0; f < HIDR_; ++f) e[f] = er[f];
  for (int a = 0; a < HIDA_; ++a) {
    float q = wq_b[a], k = wk_b[a];
    for (int f = 0; f < HIDR_; ++f) {
      q += e[f] * wq_w[f * HIDA_ + a];
      k += e[f] * wk_w[f * HIDA_ + a];
    }
    qbf[(size_t)bm * HIDA_ + a] = f2bf(q);
    kbf[(size_t)bm * HIDA_ + a] = f2bf(k);
  }
}

// ---------------- K3: WMMA attn stats -> t_scalar (B,M) ----------------
// t_scalar[b,i] = sum_j(q_i.k_j) / max(||q_i.K^T||_2, 1e-12); attn never stored.
// K(batch) = 64KB staged once per block into LDS via TDM; 8 waves share it.
__global__ __launch_bounds__(256)
void k_attn_stats(const u16* __restrict__ qbf, const u16* __restrict__ kbf,
                  float* __restrict__ tsc) {
  extern __shared__ u16 smem[];
  u16* sK = smem;                                       // M_*HIDA_ bf16 = 64KB
  const int wave = threadIdx.x >> 5;
  const int lane = threadIdx.x & 31;
  const int tile = blockIdx.x * 8 + wave;
  const int b = tile >> 6;
  const int i0 = (tile & 63) << 4;
  STAGE_LDS(sK, kbf + (size_t)b * M_ * HIDA_, M_ * HIDA_ / 4);   // 8192 x 8B
  __syncthreads();
  const v16bf aq = load_a_tile(qbf + ((size_t)b * M_ + i0) * HIDA_, HIDA_);
  float s1[8] = {0}, s2[8] = {0};
  for (int jt = 0; jt < M_ / 16; ++jt) {
    const v16bf bk = load_a_tile(sK + jt * 16 * HIDA_, HIDA_);   // LDS
    v8f c = {};
    c = wmma_bf16(aq, bk, c);                       // 16x16 attn tile
#pragma unroll
    for (int r = 0; r < 8; ++r) { const float v = c[r]; s1[r] += v; s2[r] += v * v; }
  }
#pragma unroll
  for (int r = 0; r < 8; ++r) {
    float v1 = s1[r], v2 = s2[r];
    for (int msk = 1; msk < 16; msk <<= 1) {        // reduce over N within half
      v1 += __shfl_xor(v1, msk, 32);
      v2 += __shfl_xor(v2, msk, 32);
    }
    if ((lane & 15) == r) {
      const int orow = i0 + r + 8 * ((lane >> 4) & 1);
      tsc[(size_t)b * M_ + orow] = v1 / fmaxf(sqrtf(v2), 1e-12f);
    }
  }
}

// ---------------- K4: feat (into hcat slice 3) + nv1,nv2 (bf16, K-pad 64) ---
__global__ __launch_bounds__(256)
void k_feat_nv(const float* __restrict__ emb, const float* __restrict__ tsc,
               const float* __restrict__ dvec,
               const float* __restrict__ tenc_w, const float* __restrict__ tenc_b,
               const float* __restrict__ senc_w, const float* __restrict__ senc_b,
               const float* __restrict__ gl1_w, const float* __restrict__ gl1_b,
               const float* __restrict__ gl2_w, const float* __restrict__ gl2_b,
               float* __restrict__ hcat, u16* __restrict__ nv1, u16* __restrict__ nv2) {
  const int bm = blockIdx.x * blockDim.x + threadIdx.x;
  if (bm >= B_ * M_) return;
  const int m = bm % M_;
  const float t = tsc[bm], d = dvec[m];
  float f[HIDR_];
  float* hrow = hcat + (size_t)bm * (HIDR_ * (NL_ + 1)) + NL_ * HIDR_;  // slice 3
  for (int j = 0; j < HIDR_; ++j) {
    f[j] = emb[(size_t)bm * HIDR_ + j] + t * tenc_w[j] + tenc_b[j]
           + d * senc_w[j] + senc_b[j];
    hrow[j] = f[j];
  }
  for (int j = 0; j < HIDR_; ++j) {
    float v1 = gl1_b[j], v2 = gl2_b[j];
    for (int ff = 0; ff < HIDR_; ++ff) {
      v1 += f[ff] * gl1_w[ff * HIDR_ + j];
      v2 += f[ff] * gl2_w[ff * HIDR_ + j];
    }
    nv1[(size_t)bm * 64 + j] = f2bf(tanhf(v1));
    nv2[(size_t)bm * 64 + j] = f2bf(tanhf(v2));
  }
  for (int j = HIDR_; j < 64; ++j) {                 // zero-pad K to 64
    nv1[(size_t)bm * 64 + j] = 0;
    nv2[(size_t)bm * 64 + j] = 0;
  }
}

// ---------------- K5: WMMA mask = (nv1.nv2^T - nv2.nv1^T > 0) | adj ---------
// nv1(batch)+nv2(batch) = 256KB staged into LDS (320KB/WGP on CDNA5) via TDM.
__global__ __launch_bounds__(256)
void k_mask(const u16* __restrict__ nv1, const u16* __restrict__ nv2,
            const u16* __restrict__ adjbits, u16* __restrict__ mask16) {
  extern __shared__ u16 smem[];
  u16* s1v = smem;                  // M_*64 bf16 = 128KB
  u16* s2v = smem + M_ * 64;        // 128KB
  const int wave = threadIdx.x >> 5;
  const int lane = threadIdx.x & 31;
  const int tile = blockIdx.x * 8 + wave;
  const int b = tile >> 6;
  const int i0 = (tile & 63) << 4;
  const size_t bm = (size_t)b * M_;
  STAGE_LDS(s1v, nv1 + bm * 64, M_ * 64 / 4);        // 16384 x 8B
  STAGE_LDS(s2v, nv2 + bm * 64, M_ * 64 / 4);
  __syncthreads();
  const v16bf a1lo = load_a_tile(s1v + (size_t)i0 * 64, 64);
  const v16bf a1hi = load_a_tile(s1v + (size_t)i0 * 64 + 32, 64);
  const v16bf a2lo = load_a_tile(s2v + (size_t)i0 * 64, 64);
  const v16bf a2hi = load_a_tile(s2v + (size_t)i0 * 64 + 32, 64);
  for (int jt = 0; jt < M_ / 16; ++jt) {
    const int j0 = jt << 4;
    const v16bf b2lo = load_a_tile(s2v + (size_t)j0 * 64, 64);
    const v16bf b2hi = load_a_tile(s2v + (size_t)j0 * 64 + 32, 64);
    const v16bf b1lo = load_a_tile(s1v + (size_t)j0 * 64, 64);
    const v16bf b1hi = load_a_tile(s1v + (size_t)j0 * 64 + 32, 64);
    v8f c1 = {}; c1 = wmma_bf16(a1lo, b2lo, c1); c1 = wmma_bf16(a1hi, b2hi, c1);
    v8f c2 = {}; c2 = wmma_bf16(a2lo, b1lo, c2); c2 = wmma_bf16(a2hi, b1hi, c2);
#pragma unroll
    for (int r = 0; r < 8; ++r) {
      const u32 bal = (u32)__ballot((c1[r] - c2[r]) > 0.f);   // la > 0
      if (lane == r) {
        const int orow = i0 + r;
        mask16[(bm + orow) * (M_ / 16) + jt] =
            (u16)((bal & 0xffffu) | adjbits[orow * (M_ / 16) + jt]);
      } else if (lane == 16 + r) {
        const int orow = i0 + r + 8;
        mask16[(bm + orow) * (M_ / 16) + jt] =
            (u16)((bal >> 16) | adjbits[orow * (M_ / 16) + jt]);
      }
    }
  }
}

// ---------------- per-GAT-layer kernels ----------------
// h = ns@W (40x40), e1=h.a1, e2=h.a2, Ht[b][f][m] bf16 (feature-major, pad 48)
__global__ __launch_bounds__(256)
void k_gat_pre(const float* __restrict__ hcat, const float* __restrict__ W,
               const float* __restrict__ a_vec, int in_slice,
               u16* __restrict__ Ht, float* __restrict__ e1, float* __restrict__ e2) {
  const int bm = blockIdx.x * blockDim.x + threadIdx.x;
  if (bm >= B_ * M_) return;
  const int b = bm / M_, m = bm % M_;
  const float* ns = hcat + (size_t)bm * (HIDR_ * (NL_ + 1)) + in_slice * HIDR_;
  float nsr[HIDR_];
  for (int f = 0; f < HIDR_; ++f) nsr[f] = ns[f];
  float s1 = 0.f, s2 = 0.f;
  for (int j = 0; j < HIDR_; ++j) {
    float h = 0.f;
    for (int f = 0; f < HIDR_; ++f) h += nsr[f] * W[f * HIDR_ + j];
    s1 += h * a_vec[j];
    s2 += h * a_vec[HIDR_ + j];
    Ht[((size_t)b * 48 + j) * M_ + m] = f2bf(h);
  }
  for (int j = HIDR_; j < 48; ++j) Ht[((size_t)b * 48 + j) * M_ + m] = 0;
  e1[bm] = s1;
  e2[bm] = s2;
}

// per-row masked online-softmax stats (max, sum of exp)
__global__ __launch_bounds__(256)
void k_gat_stats(const float* __restrict__ e1, const float* __restrict__ e2,
                 const u16* __restrict__ mask16,
                 float* __restrict__ mrow, float* __restrict__ srow) {
  const int bm = blockIdx.x * blockDim.x + threadIdx.x;
  if (bm >= B_ * M_) return;
  const int b = bm / M_;
  const float e1r = e1[bm];
  const float* e2b = e2 + (size_t)b * M_;
  const u32* mw = (const u32*)(mask16 + (size_t)bm * (M_ / 16));
  float mx = -3.0e38f, s = 0.f;
  for (int jw = 0; jw < M_ / 32; ++jw) {
    u32 bits = mw[jw];
    while (bits) {
      const int e = __ffs((int)bits) - 1;
      bits &= bits - 1;
      const int j = jw * 32 + e;
      float v = e1r + e2b[j];
      v = v > 0.f ? v : 0.2f * v;                    // leaky_relu(0.2)
      if (v > mx) { s = s * __expf(mx - v) + 1.f; mx = v; }
      else        { s += __expf(v - mx); }
    }
  }
  mrow[bm] = mx;
  srow[bm] = s;
}

// WMMA aggregation: ns_out = elu( softmax(e) @ h ), into hcat slice.
// H^T(batch) = 96KB + e2(batch) = 4KB staged once per block into LDS via TDM.
__global__ __launch_bounds__(256)
void k_gat_agg(const u16* __restrict__ Ht, const float* __restrict__ e1,
               const float* __restrict__ e2, const float* __restrict__ mrow,
               const float* __restrict__ srow, const u16* __restrict__ mask16,
               float* __restrict__ hcat, int out_slice) {
  extern __shared__ u16 smem[];
  u16*   sH  = smem;                              // 48*M_ bf16 = 96KB
  float* se2 = (float*)(smem + 48 * M_);          // M_ floats = 4KB
  const int wave = threadIdx.x >> 5;
  const int lane = threadIdx.x & 31;
  const int tile = blockIdx.x * 8 + wave;
  const int b = tile >> 6;
  const int i0 = (tile & 63) << 4;
  const size_t bm = (size_t)b * M_;
  STAGE_LDS(sH, Ht + (size_t)b * 48 * M_, 48 * M_ / 4);    // 12288 x 8B
  for (int j = threadIdx.x; j < M_; j += 256) se2[j] = e2[bm + j];
  __syncthreads();

  const int rlo = lane & 15;
  const int half = (lane >> 4) & 1;
  const int kb = half ? 8 : 0;
  const int row = i0 + rlo;                       // A-operand row for this lane
  const float e1r = e1[bm + row];
  const float mr  = mrow[bm + row];
  float invs[8];
#pragma unroll
  for (int r = 0; r < 8; ++r) invs[r] = 1.f / srow[bm + i0 + r + 8 * half];
  v8f acc0 = {}, acc1 = {}, acc2 = {};
  const u32* mwrow = (const u32*)(mask16 + (bm + row) * (M_ / 16));
  for (int jt = 0; jt < M_ / 32; ++jt) {
    const int j0 = jt << 5;
    const u32 mw = mwrow[jt];                     // mask bits j0..j0+31
    v16bf ap;
#pragma unroll
    for (int e = 0; e < 16; ++e) {
      const int kk = (e < 8) ? (kb + e) : (kb + 8 + e);
      const int j = j0 + kk;
      float v = e1r + se2[j];
      v = v > 0.f ? v : 0.2f * v;
      const float pv = ((mw >> kk) & 1u) ? __expf(v - mr) : 0.f;
      ap[e] = (__bf16)pv;                         // P row, unnormalized
    }
    const u16* hb = sH + j0;                      // feature-major H^T in LDS
    const v16bf b0 = load_a_tile(hb + 0 * 16 * M_, M_);
    const v16bf b1 = load_a_tile(hb + 1 * 16 * M_, M_);
    const v16bf b2 = load_a_tile(hb + 2 * 16 * M_, M_);
    acc0 = wmma_bf16(ap, b0, acc0);
    acc1 = wmma_bf16(ap, b1, acc1);
    acc2 = wmma_bf16(ap, b2, acc2);
  }
  // normalize by softmax sum, elu, scatter into hcat slice
#pragma unroll
  for (int r = 0; r < 8; ++r) {
    const int orow = i0 + r + 8 * half;
    const size_t base = (bm + orow) * (HIDR_ * (NL_ + 1)) + out_slice * HIDR_;
    float v;
    v = acc0[r] * invs[r]; v = v > 0.f ? v : (__expf(v) - 1.f); hcat[base + rlo] = v;
    v = acc1[r] * invs[r]; v = v > 0.f ? v : (__expf(v) - 1.f); hcat[base + 16 + rlo] = v;
    if (rlo < 8) {
      v = acc2[r] * invs[r]; v = v > 0.f ? v : (__expf(v) - 1.f);
      hcat[base + 32 + rlo] = v;
    }
  }
}

// ---------------- K_out: out = hcat @ out_w + out_b ----------------
__global__ __launch_bounds__(256)
void k_out(const float* __restrict__ hcat, const float* __restrict__ out_w,
           const float* __restrict__ out_b, float* __restrict__ out) {
  const int bm = blockIdx.x * blockDim.x + threadIdx.x;
  if (bm >= B_ * M_) return;
  const float* hr = hcat + (size_t)bm * (HIDR_ * (NL_ + 1));
  for (int t = 0; t < TOUT_; ++t) {
    float s = out_b[t];
    for (int f = 0; f < HIDR_ * (NL_ + 1); ++f) s += hr[f] * out_w[f * TOUT_ + t];
    out[(size_t)bm * TOUT_ + t] = s;
  }
}

// ---------------- launch ----------------
extern "C" void kernel_launch(void* const* d_in, const int* in_sizes, int n_in,
                              void* d_out, int out_size, void* d_ws, size_t ws_size,
                              hipStream_t stream) {
  (void)in_sizes; (void)n_in; (void)out_size; (void)ws_size;
  const float* x   = (const float*)d_in[0];
  const float* adj = (const float*)d_in[1];
  ConvP cp[5];
  int idx = 2;
  for (int i = 0; i < 5; ++i) {             // leaves: w,b,g,be,mu,var per branch
    cp[i].w   = (const float*)d_in[idx++];
    cp[i].b   = (const float*)d_in[idx++];
    cp[i].g   = (const float*)d_in[idx++];
    cp[i].be  = (const float*)d_in[idx++];
    cp[i].mu  = (const float*)d_in[idx++];
    cp[i].var = (const float*)d_in[idx++];
  }
  const float* wq_w   = (const float*)d_in[32];
  const float* wq_b   = (const float*)d_in[33];
  const float* wk_w   = (const float*)d_in[34];
  const float* wk_b   = (const float*)d_in[35];
  const float* tenc_w = (const float*)d_in[36];
  const float* tenc_b = (const float*)d_in[37];
  const float* senc_w = (const float*)d_in[38];
  const float* senc_b = (const float*)d_in[39];
  /* d_gate (d_in[40]) unused: sigmoid(.)>0 always, so A>0 mask == (la>0)|(adj>0) */
  const float* gl1_w  = (const float*)d_in[41];
  const float* gl1_b  = (const float*)d_in[42];
  const float* gl2_w  = (const float*)d_in[43];
  const float* gl2_b  = (const float*)d_in[44];
  const float* gat_w  = (const float*)d_in[45];
  const float* gat_a  = (const float*)d_in[46];
  const float* out_w  = (const float*)d_in[47];
  const float* out_b  = (const float*)d_in[48];
  float* out = (float*)d_out;

  // -------- workspace carve (~48 MB total, assumed <= ws_size) --------
  char* p = (char*)d_ws;
  auto carve = [&](size_t bytes) -> char* {
    char* r = p; p += (bytes + 255) & ~(size_t)255; return r;
  };
  float* emb     = (float*)carve((size_t)B_ * M_ * HIDR_ * 4);
  u16*   qbf     = (u16*)  carve((size_t)B_ * M_ * HIDA_ * 2);
  u16*   kbf     = (u16*)  carve((size_t)B_ * M_ * HIDA_ * 2);
  float* tsc     = (float*)carve((size_t)B_ * M_ * 4);
  float* dvec    = (float*)carve((size_t)M_ * 4);
  u16*   adjbits = (u16*)  carve((size_t)M_ * (M_ / 16) * 2);
  u16*   nv1     = (u16*)  carve((size_t)B_ * M_ * 64 * 2);
  u16*   nv2     = (u16*)  carve((size_t)B_ * M_ * 64 * 2);
  u16*   mask16  = (u16*)  carve((size_t)B_ * M_ * (M_ / 16) * 2);
  float* hcat    = (float*)carve((size_t)B_ * M_ * HIDR_ * (NL_ + 1) * 4);
  float* e1b     = (float*)carve((size_t)B_ * M_ * 4);
  float* e2b     = (float*)carve((size_t)B_ * M_ * 4);
  float* mrow    = (float*)carve((size_t)B_ * M_ * 4);
  float* srow    = (float*)carve((size_t)B_ * M_ * 4);
  u16*   Ht      = (u16*)  carve((size_t)B_ * 48 * M_ * 2);

  const int nBM = B_ * M_;                    // 32768
  const int tiles = B_ * (M_ / 16);           // 2048 16-row tiles
  const int wmmaBlocks = tiles / 8;           // 8 waves (256 thr) per block
  const size_t ldsAttn = (size_t)M_ * HIDA_ * 2;            // 64KB
  const size_t ldsMask = (size_t)2 * M_ * 64 * 2;           // 256KB (<=320KB WGP)
  const size_t ldsAgg  = (size_t)48 * M_ * 2 + M_ * 4;      // 96KB + 4KB

  k_emb<<<nBM / 128, 128, 0, stream>>>(x, cp[0], cp[1], cp[2], cp[3], cp[4], emb);
  k_degree<<<M_ / 256, 256, 0, stream>>>(adj, dvec, adjbits);
  k_qk<<<nBM / 256, 256, 0, stream>>>(emb, wq_w, wq_b, wk_w, wk_b, qbf, kbf);
  k_attn_stats<<<wmmaBlocks, 256, ldsAttn, stream>>>(qbf, kbf, tsc);
  k_feat_nv<<<nBM / 256, 256, 0, stream>>>(emb, tsc, dvec, tenc_w, tenc_b,
                                           senc_w, senc_b, gl1_w, gl1_b,
                                           gl2_w, gl2_b, hcat, nv1, nv2);
  k_mask<<<wmmaBlocks, 256, ldsMask, stream>>>(nv1, nv2, adjbits, mask16);

  for (int l = 0; l < NL_; ++l) {
    const int in_slice = (l == 0) ? NL_ : (l - 1);   // feat, ns1, ns2
    k_gat_pre<<<nBM / 256, 256, 0, stream>>>(hcat, gat_w + l * HIDR_ * HIDR_,
                                             gat_a + l * 2 * HIDR_, in_slice,
                                             Ht, e1b, e2b);
    k_gat_stats<<<nBM / 256, 256, 0, stream>>>(e1b, e2b, mask16, mrow, srow);
    k_gat_agg<<<wmmaBlocks, 256, ldsAgg, stream>>>(Ht, e1b, e2b, mrow, srow,
                                                   mask16, hcat, l);
  }
  k_out<<<nBM / 256, 256, 0, stream>>>(hcat, out_w, out_b, out);
}
